// StructureEmbedding_28329604285244
// MI455X (gfx1250) — matrix operand order repaired and checked by reading, not compiled, not run on hardware
//
#include <hip/hip_runtime.h>

typedef __attribute__((ext_vector_type(16))) _Float16 v16h;
typedef __attribute__((ext_vector_type(8)))  float    v8f;

#define DNE 128
#define NUM_RBF 50
#define KPAD 64
#define K1DIM 256
#define K2DIM 128
#define R_CUT 5.0f
// start = exp(-5), beta = (2/50*(1-start))^-2, mean_k = start + k*(1-start)/49
#define RBF_START 0.006737946999085467f
#define RBF_STEP  0.020270654142876f
#define RBF_BETA  633.5085f
#define PI_F 3.14159265358979f

__device__ __forceinline__ float siluf(float x) { return x / (1.0f + __expf(-x)); }

__device__ __forceinline__ float rbf_val(float er, int k) {
    if (k >= NUM_RBF) return 0.0f;
    float d = er - (RBF_START + (float)k * RBF_STEP);
    return __expf(-RBF_BETA * d * d);
}

// A-fragment K pattern (contiguous: k = 16*hi + hx, matching B loads)
__device__ __forceinline__ int a_kbase(int v2, int hi) {
    return 2 * v2 + 16 * hi;
}

// Opaque zero: defeats loop-invariant hoisting of B-fragment loads (which
// otherwise spills ~128 VGPRs of hoisted fragments to scratch).
__device__ __forceinline__ int opaque_zero() {
    int z = 0;
    asm volatile("" : "+v"(z));
    return z;
}

// ---------------------------------------------------------------------------
// Kernel 0: pack W_nrd / W_nru as f16, transposed [col][k] for B fragments.
// ---------------------------------------------------------------------------
__global__ void pack_weights_kernel(const float* __restrict__ Wnrd,
                                    const float* __restrict__ Wnru,
                                    _Float16* __restrict__ pW1,
                                    _Float16* __restrict__ pW2) {
    int total1 = DNE * K1DIM;
    int total2 = DNE * K2DIM;
    for (int idx = blockIdx.x * blockDim.x + threadIdx.x; idx < total1 + total2;
         idx += gridDim.x * blockDim.x) {
        if (idx < total1) {
            int col = idx >> 8, k = idx & 255;
            pW1[idx] = (_Float16)Wnrd[k * DNE + col];
        } else {
            int i2 = idx - total1;
            int col = i2 >> 7, k = i2 & 127;
            pW2[i2] = (_Float16)Wnru[k * DNE + col];
        }
    }
}

// ---------------------------------------------------------------------------
// Kernel 1: na = silu(z@A_na + b), nbr = silu(z@A_nbr + b), zero m. (K=25: VALU)
// ---------------------------------------------------------------------------
__global__ void node_embed_kernel(const float* __restrict__ z,
                                  const float* __restrict__ Ana_w, const float* __restrict__ Ana_b,
                                  const float* __restrict__ Anbr_w, const float* __restrict__ Anbr_b,
                                  float* __restrict__ na, float* __restrict__ nbr,
                                  float* __restrict__ m, int N) {
    long total = (long)N * DNE;
    for (long idx = (long)blockIdx.x * blockDim.x + threadIdx.x; idx < total;
         idx += (long)gridDim.x * blockDim.x) {
        int node = (int)(idx >> 7);
        int col  = (int)(idx & 127);
        float s1 = Ana_b[col];
        float s2 = Anbr_b[col];
        const float* zp = z + (long)node * 25;
#pragma unroll
        for (int k = 0; k < 25; ++k) {
            float zv = zp[k];
            s1 = fmaf(zv, Ana_w[k * DNE + col], s1);
            s2 = fmaf(zv, Anbr_w[k * DNE + col], s2);
        }
        na[idx]  = siluf(s1);
        nbr[idx] = siluf(s2);
        m[idx]   = 0.0f;  // zero accumulator for segment_sum
    }
}

// Build the A fragments (16 edges x K=64 rbf, f16) directly in registers.
__device__ __forceinline__ void build_rbf_frags(float er, int hi, v16h& a_lo, v16h& a_hi) {
#pragma unroll
    for (int hx = 0; hx < 16; ++hx) {
        int kb = a_kbase(hx >> 1, hi) + (hx & 1);
        a_lo[hx] = (_Float16)rbf_val(er, kb);
        a_hi[hx] = (_Float16)rbf_val(er, kb + 32);
    }
}

// Convert W (NUM_RBF x 128, f32) into LDS as [col][K=64] f16 (zero padded).
__device__ __forceinline__ void load_w_to_lds(_Float16* sW, const float* __restrict__ W) {
    for (int idx = threadIdx.x; idx < DNE * KPAD; idx += blockDim.x) {
        int n = idx >> 6;
        int k = idx & 63;
        float w = (k < NUM_RBF) ? W[k * DNE + n] : 0.0f;
        sW[idx] = (_Float16)w;
    }
}

// ---------------------------------------------------------------------------
// Kernel 2: per 16-edge tile: gate = (rbf@W_ndp + b)*c ; msg = nbr[j]*gate ;
// atomicAdd into m[i]. Also writes rt0, rt1(u), rt2(sh_l2), c.
// ---------------------------------------------------------------------------
__global__ __launch_bounds__(256)
void edge_msg_kernel(const int* __restrict__ edge,
                     const float* __restrict__ r, const float* __restrict__ u,
                     const float* __restrict__ Wndp, const float* __restrict__ bndp,
                     const float* __restrict__ nbr, float* __restrict__ m,
                     float* __restrict__ rt0, float* __restrict__ rt1,
                     float* __restrict__ rt2, float* __restrict__ cout,
                     long E) {
    __shared__ _Float16 sW[DNE * KPAD];
    load_w_to_lds(sW, Wndp);
    __syncthreads();

    int lane = threadIdx.x & 31;
    int wave = threadIdx.x >> 5;
    int wpb  = blockDim.x >> 5;
    int hi   = lane >> 4;
    int lcol = lane & 15;
    long tiles = E >> 4;

    for (long tile = (long)blockIdx.x * wpb + wave; tile < tiles;
         tile += (long)gridDim.x * wpb) {
        long eb    = tile << 4;
        long e_row = eb + lcol;

        float r_e = r[e_row];
        int   i_r = edge[2 * e_row];
        int   j_r = edge[2 * e_row + 1];
        float c_r = (r_e < R_CUT) ? 0.5f * (__cosf(PI_F * r_e * (1.0f / R_CUT)) + 1.0f) : 0.0f;

        if (hi == 0) {  // cheap per-edge outputs, one lane per edge
            rt0[e_row]  = r_e;
            cout[e_row] = c_r;
            float ux = u[3 * e_row], uy = u[3 * e_row + 1], uz = u[3 * e_row + 2];
            rt1[3 * e_row]     = ux;
            rt1[3 * e_row + 1] = uy;
            rt1[3 * e_row + 2] = uz;
            const float s3 = 1.7320508075688772f;
            rt2[5 * e_row]     = s3 * ux * uy;
            rt2[5 * e_row + 1] = s3 * uy * uz;
            rt2[5 * e_row + 2] = 0.5f * (3.0f * uz * uz - 1.0f);
            rt2[5 * e_row + 3] = s3 * uz * ux;
            rt2[5 * e_row + 4] = 0.5f * s3 * (ux * ux - uy * uy);
        }

        float er = __expf(-r_e);  // alpha = 5/R_CUT = 1
        v16h a_lo, a_hi;
        build_rbf_frags(er, hi, a_lo, a_hi);

        float ce8[8]; int ie8[8], je8[8];
#pragma unroll
        for (int v = 0; v < 8; ++v) {
            int rsrc = v + 8 * hi;
            ce8[v] = __shfl(c_r, rsrc, 32);
            ie8[v] = __shfl(i_r, rsrc, 32);
            je8[v] = __shfl(j_r, rsrc, 32);
        }

        // Phase 1: all WMMAs; zoff keeps B loads per-iteration (no hoist+spill)
        int zoff = opaque_zero();
        v8f acc[8];
#pragma unroll
        for (int nt = 0; nt < 8; ++nt) {
            int col = nt * 16 + lcol;
            v16h b_lo = *(const v16h*)&sW[col * KPAD + 16 * hi + zoff];
            v16h b_hi = *(const v16h*)&sW[col * KPAD + 32 + 16 * hi + zoff];
            v8f a0 = {};
            a0 = __builtin_amdgcn_wmma_f32_16x16x32_f16(false, a_lo, false, b_lo,
                                                        (short)0, a0, false, false);
            acc[nt] = __builtin_amdgcn_wmma_f32_16x16x32_f16(false, a_hi, false, b_hi,
                                                             (short)0, a0, false, false);
        }
        // Phase 2: batched gathers then atomics per N-tile
#pragma unroll
        for (int nt = 0; nt < 8; ++nt) {
            int col = nt * 16 + lcol;
            float bias = bndp[col];
            float nv[8];
#pragma unroll
            for (int v = 0; v < 8; ++v)
                nv[v] = nbr[(long)je8[v] * DNE + col];
#pragma unroll
            for (int v = 0; v < 8; ++v) {
                float msg = (acc[nt][v] + bias) * ce8[v] * nv[v];
                atomicAdd(&m[(long)ie8[v] * DNE + col], msg);
            }
        }
    }
}

// ---------------------------------------------------------------------------
// Kernel 3: WMMA node update. One wave = 16 nodes.
//   h0 = concat(na,m) @ W_nrd + b   (K=256, 8 chunks, 64 WMMA/wave)
//   LayerNorm in-register on D fragments (shfl_xor row reduce), silu,
//   re-transpose through LDS (wave-private), then @ W_nru + b (32 WMMA/wave).
// ---------------------------------------------------------------------------
__global__ __launch_bounds__(128)
void node_update_wmma_kernel(const float* __restrict__ na, const float* __restrict__ m,
                             const _Float16* __restrict__ pW1, const float* __restrict__ bnrd,
                             const _Float16* __restrict__ pW2, const float* __restrict__ bnru,
                             const float* __restrict__ lng, const float* __restrict__ lnb,
                             float* __restrict__ hout, int N) {
    __shared__ _Float16 sX[64 * K1DIM];  // 32 KB: staged concat(na, m), f16
    __shared__ _Float16 sS[64 * K2DIM];  // 16 KB: silu(LN(h0)), f16

    int lane = threadIdx.x & 31;
    int wave = threadIdx.x >> 5;
    int hi   = lane >> 4;
    int lrow = lane & 15;
    int rbase = wave * 16;
    int myrow = rbase + lrow;

    int chunks = (N + 63) >> 6;
    for (int ch = blockIdx.x; ch < chunks; ch += gridDim.x) {
        long nb = (long)ch << 6;
        __syncthreads();  // protect sX/sS reuse across chunk iterations
        for (int idx = threadIdx.x; idx < 64 * K1DIM; idx += blockDim.x) {
            int n = idx >> 8, k = idx & 255;
            long node = nb + n;
            float vx = 0.0f;
            if (node < N) vx = (k < 128) ? na[node * DNE + k] : m[node * DNE + (k - 128)];
            sX[idx] = (_Float16)vx;
        }
        __syncthreads();

        int zoff = opaque_zero();  // keep B-fragment loads inside the loop

        // ---- matmul 1: [16 x 256] @ [256 x 128] ----
        v8f acc[8];
#pragma unroll
        for (int nt = 0; nt < 8; ++nt) acc[nt] = (v8f){};
#pragma unroll
        for (int c = 0; c < 8; ++c) {
            union { v16h h; unsigned int u[8]; } af;
#pragma unroll
            for (int v2 = 0; v2 < 8; ++v2) {
                int kk = a_kbase(v2, hi) + c * 32;
                af.u[v2] = *(const unsigned int*)&sX[myrow * K1DIM + kk];
            }
#pragma unroll
            for (int nt = 0; nt < 8; ++nt) {
                int col = nt * 16 + lrow;
                v16h b = *(const v16h*)&pW1[col * K1DIM + c * 32 + 16 * hi + zoff];
                acc[nt] = __builtin_amdgcn_wmma_f32_16x16x32_f16(false, af.h, false, b,
                                                                 (short)0, acc[nt], false, false);
            }
        }

        // ---- bias + LayerNorm on D fragments ----
        // D element (row = v + 8*hi [+rbase], col = nt*16 + lrow); for fixed v,
        // the 16 lanes of a half hold the same row -> shfl_xor row reduction.
        float bias8[8];
#pragma unroll
        for (int nt = 0; nt < 8; ++nt) bias8[nt] = bnrd[nt * 16 + lrow];
#pragma unroll
        for (int nt = 0; nt < 8; ++nt)
#pragma unroll
            for (int v = 0; v < 8; ++v) acc[nt][v] += bias8[nt];

        float mu[8], rstd[8];
#pragma unroll
        for (int v = 0; v < 8; ++v) {
            float s = 0.0f;
#pragma unroll
            for (int nt = 0; nt < 8; ++nt) s += acc[nt][v];
#pragma unroll
            for (int msk = 1; msk < 16; msk <<= 1) s += __shfl_xor(s, msk, 32);
            mu[v] = s * (1.0f / 128.0f);
        }
#pragma unroll
        for (int v = 0; v < 8; ++v) {
            float s = 0.0f;
#pragma unroll
            for (int nt = 0; nt < 8; ++nt) {
                float d = acc[nt][v] - mu[v];
                s += d * d;
            }
#pragma unroll
            for (int msk = 1; msk < 16; msk <<= 1) s += __shfl_xor(s, msk, 32);
            rstd[v] = rsqrtf(s * (1.0f / 128.0f) + 1e-5f);
        }

        // normalize + silu -> sS (wave-private rows; no block barrier needed)
#pragma unroll
        for (int nt = 0; nt < 8; ++nt) {
            int col = nt * 16 + lrow;
            float g = lng[col], bb = lnb[col];
#pragma unroll
            for (int v = 0; v < 8; ++v) {
                int rowl = rbase + v + 8 * hi;
                float hn = (acc[nt][v] - mu[v]) * rstd[v] * g + bb;
                sS[rowl * K2DIM + col] = (_Float16)siluf(hn);
            }
        }

        // ---- matmul 2: [16 x 128] @ [128 x 128] ----
        v8f acc2[8];
#pragma unroll
        for (int nt = 0; nt < 8; ++nt) acc2[nt] = (v8f){};
#pragma unroll
        for (int c = 0; c < 4; ++c) {
            union { v16h h; unsigned int u[8]; } af;
#pragma unroll
            for (int v2 = 0; v2 < 8; ++v2) {
                int kk = a_kbase(v2, hi) + c * 32;
                af.u[v2] = *(const unsigned int*)&sS[myrow * K2DIM + kk];
            }
#pragma unroll
            for (int nt = 0; nt < 8; ++nt) {
                int col = nt * 16 + lrow;
                v16h b = *(const v16h*)&pW2[col * K2DIM + c * 32 + 16 * hi + zoff];
                acc2[nt] = __builtin_amdgcn_wmma_f32_16x16x32_f16(false, af.h, false, b,
                                                                  (short)0, acc2[nt], false, false);
            }
        }

        // ---- write h ----
#pragma unroll
        for (int nt = 0; nt < 8; ++nt) {
            int col = nt * 16 + lrow;
            float bo = bnru[col];
#pragma unroll
            for (int v = 0; v < 8; ++v) {
                long node = nb + rbase + v + 8 * hi;
                if (node < N) hout[node * DNE + col] = acc2[nt][v] + bo;
            }
        }
    }
}

// ---------------------------------------------------------------------------
// Kernel 4: t = (h[i] + h[j]) * (rbf @ W_erp + b)   (same WMMA tiling)
// ---------------------------------------------------------------------------
__global__ __launch_bounds__(256)
void edge_out_kernel(const int* __restrict__ edge, const float* __restrict__ r,
                     const float* __restrict__ Werp, const float* __restrict__ berp,
                     const float* __restrict__ h, float* __restrict__ t,
                     long E) {
    __shared__ _Float16 sW[DNE * KPAD];
    load_w_to_lds(sW, Werp);
    __syncthreads();

    int lane = threadIdx.x & 31;
    int wave = threadIdx.x >> 5;
    int wpb  = blockDim.x >> 5;
    int hi   = lane >> 4;
    int lcol = lane & 15;
    long tiles = E >> 4;

    for (long tile = (long)blockIdx.x * wpb + wave; tile < tiles;
         tile += (long)gridDim.x * wpb) {
        long eb    = tile << 4;
        long e_row = eb + lcol;

        float r_e = r[e_row];
        int   i_r = edge[2 * e_row];
        int   j_r = edge[2 * e_row + 1];

        float er = __expf(-r_e);
        v16h a_lo, a_hi;
        build_rbf_frags(er, hi, a_lo, a_hi);

        int ie8[8], je8[8];
#pragma unroll
        for (int v = 0; v < 8; ++v) {
            int rsrc = v + 8 * hi;
            ie8[v] = __shfl(i_r, rsrc, 32);
            je8[v] = __shfl(j_r, rsrc, 32);
        }

        int zoff = opaque_zero();
        v8f acc[8];
#pragma unroll
        for (int nt = 0; nt < 8; ++nt) {
            int col = nt * 16 + lcol;
            v16h b_lo = *(const v16h*)&sW[col * KPAD + 16 * hi + zoff];
            v16h b_hi = *(const v16h*)&sW[col * KPAD + 32 + 16 * hi + zoff];
            v8f a0 = {};
            a0 = __builtin_amdgcn_wmma_f32_16x16x32_f16(false, a_lo, false, b_lo,
                                                        (short)0, a0, false, false);
            acc[nt] = __builtin_amdgcn_wmma_f32_16x16x32_f16(false, a_hi, false, b_hi,
                                                             (short)0, a0, false, false);
        }
#pragma unroll
        for (int nt = 0; nt < 8; ++nt) {
            int col = nt * 16 + lcol;
            float bias = berp[col];
            float hv[8];
#pragma unroll
            for (int v = 0; v < 8; ++v)
                hv[v] = h[(long)ie8[v] * DNE + col] + h[(long)je8[v] * DNE + col];
#pragma unroll
            for (int v = 0; v < 8; ++v) {
                long e = eb + v + 8 * hi;
                t[e * DNE + col] = hv[v] * (acc[nt][v] + bias);
            }
        }
    }
}

// ---------------------------------------------------------------------------
extern "C" void kernel_launch(void* const* d_in, const int* in_sizes, int n_in,
                              void* d_out, int out_size, void* d_ws, size_t ws_size,
                              hipStream_t stream) {
    const float* z      = (const float*)d_in[0];
    const int*   edge   = (const int*)d_in[1];
    const float* r      = (const float*)d_in[2];
    const float* u      = (const float*)d_in[3];
    const float* Ana_w  = (const float*)d_in[4];
    const float* Ana_b  = (const float*)d_in[5];
    const float* Anbr_w = (const float*)d_in[6];
    const float* Anbr_b = (const float*)d_in[7];
    const float* Wndp_w = (const float*)d_in[8];
    const float* Wndp_b = (const float*)d_in[9];
    const float* Wnrd_w = (const float*)d_in[10];
    const float* Wnrd_b = (const float*)d_in[11];
    const float* Wnru_w = (const float*)d_in[12];
    const float* Wnru_b = (const float*)d_in[13];
    const float* Werp_w = (const float*)d_in[14];
    const float* Werp_b = (const float*)d_in[15];
    const float* ln_g   = (const float*)d_in[16];
    const float* ln_b   = (const float*)d_in[17];

    int  N = in_sizes[0] / 25;
    long E = (long)in_sizes[2];

    float* out_h   = (float*)d_out;
    float* out_t   = out_h + (long)N * DNE;
    float* out_rt0 = out_t + E * DNE;
    float* out_rt1 = out_rt0 + E;
    float* out_rt2 = out_rt1 + 3 * E;
    float* out_c   = out_rt2 + 5 * E;

    float*    ws_na  = (float*)d_ws;
    float*    ws_nbr = ws_na + (long)N * DNE;
    float*    ws_m   = ws_nbr + (long)N * DNE;
    _Float16* ws_pW1 = (_Float16*)(ws_m + (long)N * DNE);
    _Float16* ws_pW2 = ws_pW1 + DNE * K1DIM;

    // K0: pack f16-transposed weights for kernel 3's B fragments (lives in L2)
    pack_weights_kernel<<<96, 256, 0, stream>>>(Wnrd_w, Wnru_w, ws_pW1, ws_pW2);

    // K1: node embeddings + zero m
    {
        long total = (long)N * DNE;
        int blocks = (int)((total + 255) / 256);
        if (blocks > 25000) blocks = 25000;
        node_embed_kernel<<<blocks, 256, 0, stream>>>(z, Ana_w, Ana_b, Anbr_w, Anbr_b,
                                                      ws_na, ws_nbr, ws_m, N);
    }
    // K2: edge messages (WMMA) + scatter + rt outputs
    edge_msg_kernel<<<4096, 256, 0, stream>>>(edge, r, u, Wndp_w, Wndp_b, ws_nbr, ws_m,
                                              out_rt0, out_rt1, out_rt2, out_c, E);
    // K3: node update (WMMA matmul + in-register LN + silu + WMMA matmul)
    {
        int blocks = (N + 63) / 64;
        node_update_wmma_kernel<<<blocks, 128, 0, stream>>>(ws_na, ws_m, ws_pW1, Wnrd_b,
                                                            ws_pW2, Wnru_b, ln_g, ln_b,
                                                            out_h, N);
    }
    // K4: edge outputs t (WMMA)
    edge_out_kernel<<<4096, 256, 0, stream>>>(edge, r, Werp_w, Werp_b, out_h, out_t, E);
}
